// GAT_58231166599311
// MI455X (gfx1250) — matrix operand order repaired and checked by reading, not compiled, not run on hardware
//
#include <hip/hip_runtime.h>
#include <hip/hip_bf16.h>
#include <math.h>

typedef __attribute__((ext_vector_type(16))) _Float16 v16h;
typedef __attribute__((ext_vector_type(8)))  _Float16 v8h;
typedef __attribute__((ext_vector_type(8)))  float    v8f;

#define NU 50000
#define NM 10000
#define NE 250000

__device__ __forceinline__ v16h cat8(v8h lo, v8h hi) {
  return __builtin_shufflevector(lo, hi, 0,1,2,3,4,5,6,7,8,9,10,11,12,13,14,15);
}

// ---------------------------------------------------------------------------
// Tiled GEMM: C[M,N] = act(A[M,K] @ W[K,N] + bias) using v_wmma_f32_16x16x32_f16
// block = 256 threads (8 waves), 64x64 output tile, wave = 16x32 strip.
// A/B staged through LDS as f16; fragments read as 16B ds_load_b128.
// Requires: K % 32 == 0, N % 64 == 0 (holds for all shapes here).
// ---------------------------------------------------------------------------
__global__ __launch_bounds__(256)
void gemm_wmma_f16(const float* __restrict__ A, const float* __restrict__ W,
                   const float* __restrict__ bias, float* __restrict__ C,
                   int M, int N, int K, int relu) {
  __shared__ __align__(16) _Float16 ldsA[64 * 32];
  __shared__ __align__(16) _Float16 ldsB[64 * 32];   // transposed: [n][k]

  const int m0 = blockIdx.x * 64;
  const int n0 = blockIdx.y * 64;
  const int t    = threadIdx.x;
  const int lane = t & 31;
  const int wave = t >> 5;
  const int wm = wave & 3;        // 0..3 -> M sub-tile
  const int wn = wave >> 2;       // 0..1 -> N half (32 cols)
  const int r  = lane & 15;
  const int h  = lane >> 4;

  v8f c0 = {}, c1 = {};

  for (int k0 = 0; k0 < K; k0 += 32) {
    // --- stage A tile (64 rows x 32 K), fp32 -> f16 ---
    {
      int row = t >> 2;
      int kk  = (t & 3) * 8;
      int gr  = m0 + row; if (gr >= M) gr = M - 1;   // clamp (stores are guarded)
      const float* s = A + (size_t)gr * K + k0 + kk;
      _Float16* dA = &ldsA[row * 32 + kk];
      #pragma unroll
      for (int i = 0; i < 8; ++i) dA[i] = (_Float16)s[i];
    }
    // --- stage B tile transposed: ldsB[n][k] = W[k0+k][n0+n] ---
    {
      int kp = t >> 4;            // k pair 0..15
      int nb = (t & 15) * 4;      // 4 columns
      const float* s0 = W + (size_t)(k0 + 2 * kp) * N + n0 + nb;
      const float* s1 = s0 + N;
      #pragma unroll
      for (int i = 0; i < 4; ++i) {
        ldsB[(nb + i) * 32 + 2 * kp]     = (_Float16)s0[i];
        ldsB[(nb + i) * 32 + 2 * kp + 1] = (_Float16)s1[i];
      }
    }
    // prefetch next K tile while this one is consumed
    if (k0 + 32 < K) {
      int row = t >> 2;
      int gr  = m0 + row; if (gr >= M) gr = M - 1;
      __builtin_prefetch(A + (size_t)gr * K + k0 + 32 + (t & 3) * 8, 0, 1);
      __builtin_prefetch(W + (size_t)(k0 + 32 + (t >> 4) * 2) * N + n0 + (t & 15) * 4, 0, 1);
    }
    __syncthreads();

    // --- fragments ---
    // A (16x32 f16): lanes 0-15 K=0..15, lanes 16-31 K=16..31 -> contiguous 16 K at h*16
    const v8h* pa = (const v8h*)&ldsA[(wm * 16 + r) * 32 + h * 16];
    v16h av = cat8(pa[0], pa[1]);
    // B (32x16 f16): lanes 0-15 K=0..15, lanes 16-31 K=16..31 -> contiguous 16 K at h*16
    const v8h* pb0 = (const v8h*)&ldsB[(wn * 32 + r) * 32 + h * 16];
    v16h bv0 = cat8(pb0[0], pb0[1]);
    const v8h* pb1 = (const v8h*)&ldsB[(wn * 32 + 16 + r) * 32 + h * 16];
    v16h bv1 = cat8(pb1[0], pb1[1]);

    c0 = __builtin_amdgcn_wmma_f32_16x16x32_f16(false, av, false, bv0, (short)0, c0, false, false);
    c1 = __builtin_amdgcn_wmma_f32_16x16x32_f16(false, av, false, bv1, (short)0, c1, false, false);
    __syncthreads();
  }

  // --- store: C/D layout: lane N = lane&15; VGPR v -> row v + 8*(lane>>4) ---
  const int cn = lane & 15;
  const int col0 = n0 + wn * 32 + cn;
  const float b0v = bias ? bias[col0]      : 0.f;
  const float b1v = bias ? bias[col0 + 16] : 0.f;
  #pragma unroll
  for (int v = 0; v < 8; ++v) {
    int mrow = m0 + wm * 16 + v + 8 * h;
    if (mrow < M) {
      float x0 = c0[v] + b0v;
      float x1 = c1[v] + b1v;
      if (relu) { x0 = fmaxf(x0, 0.f); x1 = fmaxf(x1, 0.f); }
      C[(size_t)mrow * N + col0]      = x0;
      C[(size_t)mrow * N + col0 + 16] = x1;
    }
  }
}

// ---------------------------------------------------------------------------
// Edge-phase kernels (GATv2 attention + segment softmax + scatter)
// xl/xr layout: [N, H=2, C=128] row-major (stride 256)
// ---------------------------------------------------------------------------
__device__ __forceinline__ void atomicMaxF(float* addr, float v) {
  if (v >= 0.f) atomicMax((int*)addr, __float_as_int(v));
  else          atomicMin((unsigned int*)addr, (unsigned int)__float_as_int(v));
}

// seed out with bias, reset emax/denom
__global__ void init_out_kernel(float* __restrict__ out, const float* __restrict__ bias,
                                float* __restrict__ emax, float* __restrict__ denom, int Nd) {
  int i = blockIdx.x * blockDim.x + threadIdx.x;
  if (i >= Nd * 256) return;
  int j = i & 255;
  out[i] = bias[j];
  if (j < 2) {
    int n = i >> 8;
    emax[n * 2 + j]  = -INFINITY;
    denom[n * 2 + j] = 0.f;
  }
}

// one wave per (edge, head): e = sum_c leakyrelu(xl[s]+xr[d]) * att; atomic segment max
__global__ __launch_bounds__(256)
void edge_scores_kernel(const float* __restrict__ xl, const float* __restrict__ xr,
                        const int* __restrict__ src, const int* __restrict__ dst,
                        const float* __restrict__ att,
                        float* __restrict__ ebuf, float* __restrict__ emax, int nE) {
  int task = blockIdx.x * 8 + (threadIdx.x >> 5);
  if (task >= nE * 2) return;          // wave-uniform
  int lane = threadIdx.x & 31;
  int e = task >> 1, hd = task & 1;
  int s = src[e], d = dst[e];
  const float4 l = ((const float4*)(xl + (size_t)s * 256 + hd * 128))[lane];
  const float4 rr = ((const float4*)(xr + (size_t)d * 256 + hd * 128))[lane];
  const float4 a = ((const float4*)(att + hd * 128))[lane];
  float v, acc = 0.f;
  v = l.x + rr.x; v = v > 0.f ? v : 0.2f * v; acc += v * a.x;
  v = l.y + rr.y; v = v > 0.f ? v : 0.2f * v; acc += v * a.y;
  v = l.z + rr.z; v = v > 0.f ? v : 0.2f * v; acc += v * a.z;
  v = l.w + rr.w; v = v > 0.f ? v : 0.2f * v; acc += v * a.w;
  #pragma unroll
  for (int m = 16; m > 0; m >>= 1) acc += __shfl_xor(acc, m);
  if (lane == 0) {
    ebuf[e * 2 + hd] = acc;
    atomicMaxF(&emax[d * 2 + hd], acc);
  }
}

// per (edge, head): w = exp(e - emax[dst]); atomic segment sum
__global__ void edge_exp_kernel(const float* __restrict__ ebuf, const int* __restrict__ dst,
                                const float* __restrict__ emax,
                                float* __restrict__ wbuf, float* __restrict__ denom, int nE) {
  int i = blockIdx.x * blockDim.x + threadIdx.x;
  if (i >= nE * 2) return;
  int e = i >> 1, hd = i & 1;
  int d = dst[e];
  float w = __expf(ebuf[i] - emax[d * 2 + hd]);
  wbuf[i] = w;
  atomicAdd(&denom[d * 2 + hd], w);
}

// one wave per (edge, head): out[dst] += alpha * xl[src]
__global__ __launch_bounds__(256)
void edge_accum_kernel(const float* __restrict__ xl,
                       const int* __restrict__ src, const int* __restrict__ dst,
                       const float* __restrict__ wbuf, const float* __restrict__ denom,
                       float* __restrict__ out, int nE) {
  int task = blockIdx.x * 8 + (threadIdx.x >> 5);
  if (task >= nE * 2) return;          // wave-uniform
  int lane = threadIdx.x & 31;
  int e = task >> 1, hd = task & 1;
  int s = src[e], d = dst[e];
  float alpha = wbuf[e * 2 + hd] / (denom[d * 2 + hd] + 1e-16f);
  const float4 l = ((const float4*)(xl + (size_t)s * 256 + hd * 128))[lane];
  float* po = out + (size_t)d * 256 + hd * 128 + lane * 4;
  atomicAdd(po + 0, alpha * l.x);
  atomicAdd(po + 1, alpha * l.y);
  atomicAdd(po + 2, alpha * l.z);
  atomicAdd(po + 3, alpha * l.w);
}

__global__ void relu_kernel(float* __restrict__ x, int n) {
  int i = blockIdx.x * blockDim.x + threadIdx.x;
  if (i < n) x[i] = fmaxf(x[i], 0.f);
}

// ---------------------------------------------------------------------------
// Host orchestration
// ---------------------------------------------------------------------------
static inline void run_gemm(const float* A, const float* W, const float* bias, float* C,
                            int M, int N, int K, int relu, hipStream_t s) {
  dim3 grid((M + 63) / 64, N / 64);
  gemm_wmma_f16<<<grid, 256, 0, s>>>(A, W, bias, C, M, N, K, relu);
}

static inline void run_gatv2(const float* xl, const float* xr,
                             const int* src, const int* dst, int Nd,
                             const float* att, const float* bias, float* out,
                             float* ebuf, float* wbuf, float* emax, float* denom,
                             hipStream_t s) {
  init_out_kernel<<<(Nd * 256 + 255) / 256, 256, 0, s>>>(out, bias, emax, denom, Nd);
  edge_scores_kernel<<<(2 * NE + 7) / 8, 256, 0, s>>>(xl, xr, src, dst, att, ebuf, emax, NE);
  edge_exp_kernel<<<(2 * NE + 255) / 256, 256, 0, s>>>(ebuf, dst, emax, wbuf, denom, NE);
  edge_accum_kernel<<<(2 * NE + 7) / 8, 256, 0, s>>>(xl, src, dst, wbuf, denom, out, NE);
}

extern "C" void kernel_launch(void* const* d_in, const int* in_sizes, int n_in,
                              void* d_out, int out_size, void* d_ws, size_t ws_size,
                              hipStream_t stream) {
  const float* user_f = (const float*)d_in[0];     // [NU,64]
  const float* item_f = (const float*)d_in[1];     // [NM,128]
  const int*   e_src  = (const int*)d_in[2];       // [E] user idx
  const int*   e_dst  = (const int*)d_in[3];       // [E] movie idx
  const float* up_w = (const float*)d_in[4];
  const float* up_b = (const float*)d_in[5];
  const float* ip_w = (const float*)d_in[6];
  const float* ip_b = (const float*)d_in[7];
  // layers[l][um|mu]{Wl,Wr,att,bias} flattened in dict order starting at 8
  const float* L[2][2][4];
  for (int l = 0; l < 2; ++l)
    for (int et = 0; et < 2; ++et)
      for (int p = 0; p < 4; ++p)
        L[l][et][p] = (const float*)d_in[8 + l * 8 + et * 4 + p];

  float* ws = (float*)d_ws;
  float* u0   = ws;                 ws += (size_t)NU * 128;
  float* m0   = ws;                 ws += (size_t)NM * 128;
  float* xA   = ws;                 ws += (size_t)NU * 256;  // user-side transform
  float* xB   = ws;                 ws += (size_t)NM * 256;  // movie-side transform
  float* u1   = ws;                 ws += (size_t)NU * 256;
  float* m1   = ws;                 ws += (size_t)NM * 256;
  float* ebuf = ws;                 ws += (size_t)NE * 2;
  float* wbuf = ws;                 ws += (size_t)NE * 2;
  float* emax = ws;                 ws += (size_t)NU * 2;
  float* denom= ws;                 ws += (size_t)NU * 2;

  float* out_u = (float*)d_out;             // [NU,256]
  float* out_m = out_u + (size_t)NU * 256;  // [NM,256]

  // projections
  run_gemm(user_f, up_w, up_b, u0, NU, 128, 64, 0, stream);
  run_gemm(item_f, ip_w, ip_b, m0, NM, 128, 128, 0, stream);

  // ---- layer 0 (in_dim = 128) ----
  // um: src=user, dst=movie
  run_gemm(u0, L[0][0][0], nullptr, xA, NU, 256, 128, 0, stream);   // xl = u @ Wl
  run_gemm(m0, L[0][0][1], nullptr, xB, NM, 256, 128, 0, stream);   // xr = m @ Wr
  run_gatv2(xA, xB, e_src, e_dst, NM, L[0][0][2], L[0][0][3], m1,
            ebuf, wbuf, emax, denom, stream);
  // mu: src=movie, dst=user
  run_gemm(m0, L[0][1][0], nullptr, xB, NM, 256, 128, 0, stream);   // xl = m @ Wl
  run_gemm(u0, L[0][1][1], nullptr, xA, NU, 256, 128, 0, stream);   // xr = u @ Wr
  run_gatv2(xB, xA, e_dst, e_src, NU, L[0][1][2], L[0][1][3], u1,
            ebuf, wbuf, emax, denom, stream);
  // inter-layer ReLU
  relu_kernel<<<((size_t)NU * 256 + 255) / 256, 256, 0, stream>>>(u1, NU * 256);
  relu_kernel<<<((size_t)NM * 256 + 255) / 256, 256, 0, stream>>>(m1, NM * 256);

  // ---- layer 1 (in_dim = 256), outputs straight to d_out ----
  run_gemm(u1, L[1][0][0], nullptr, xA, NU, 256, 256, 0, stream);
  run_gemm(m1, L[1][0][1], nullptr, xB, NM, 256, 256, 0, stream);
  run_gatv2(xA, xB, e_src, e_dst, NM, L[1][0][2], L[1][0][3], out_m,
            ebuf, wbuf, emax, denom, stream);
  run_gemm(m1, L[1][1][0], nullptr, xB, NM, 256, 256, 0, stream);
  run_gemm(u1, L[1][1][1], nullptr, xA, NU, 256, 256, 0, stream);
  run_gatv2(xB, xA, e_dst, e_src, NU, L[1][1][2], L[1][1][3], out_u,
            ebuf, wbuf, emax, denom, stream);
}
